// END_88751204205249
// MI455X (gfx1250) — matrix operand order repaired and checked by reading, not compiled, not run on hardware
//
#include <hip/hip_runtime.h>
#include <math.h>

typedef __attribute__((ext_vector_type(8)))  float  v8f;
typedef __attribute__((ext_vector_type(16))) __bf16 v16bf;

union Frag { v16bf v; uint4 q[2]; unsigned int u[8]; };

__device__ inline unsigned short f2bf(float x) {
  union { float f; unsigned int u; } c; c.f = x;
  unsigned int r = c.u + 0x7FFFu + ((c.u >> 16) & 1u);
  return (unsigned short)(r >> 16);
}
__device__ inline unsigned int pack2(float lo, float hi) {
  return (unsigned int)f2bf(lo) | ((unsigned int)f2bf(hi) << 16);
}

// CDNA5 async global->LDS copy (VGLOBAL, GVS mode), tracked by ASYNCcnt.
__device__ inline void async_b128(unsigned int ldsOff, unsigned int gOff, const void* sbase) {
  asm volatile("global_load_async_to_lds_b128 %0, %1, %2"
               :: "v"(ldsOff), "v"(gOff), "s"(sbase) : "memory");
}
__device__ inline void wait_async3() { asm volatile("s_wait_asynccnt 0x3" ::: "memory"); }
__device__ inline void wait_async0() { asm volatile("s_wait_asynccnt 0x0" ::: "memory"); }

// generic 3x3 inverse, same adjugate/det ordering as the reference
__device__ inline void inv3(const float* m, float* o) {
  float a=m[0],b=m[1],c=m[2],d=m[3],e=m[4],f=m[5],g=m[6],h=m[7],i=m[8];
  float det = a*e*i + b*f*g + d*h*c - g*e*c - a*h*f - d*b*i;
  float s = 1.0f / det;
  o[0]=(e*i-f*h)*s; o[1]=(c*h-b*i)*s; o[2]=(b*f-c*e)*s;
  o[3]=(f*g-d*i)*s; o[4]=(a*i-c*g)*s; o[5]=(c*d-a*f)*s;
  o[6]=(d*h-e*g)*s; o[7]=(b*g-a*h)*s; o[8]=(a*e-b*d)*s;
}
// C = V @ (iU - I)
__device__ inline void cmat(const float* V, const float* iU, float* C) {
  float A[9];
#pragma unroll
  for (int k=0;k<9;++k) A[k]=iU[k];
  A[0]-=1.f; A[4]-=1.f; A[8]-=1.f;
#pragma unroll
  for (int i=0;i<3;++i)
#pragma unroll
    for (int j=0;j<3;++j)
      C[i*3+j] = V[i*3+0]*A[j] + V[i*3+1]*A[3+j] + V[i*3+2]*A[6+j];
}

// ---------------------------------------------------------------------------
// bf16 WMMA GEMM:  C[N x M] = A[N x K] @ W[K x M]   (A bf16 row-major,
// W supplied transposed bf16 [M][K]).  256 threads = 8 waves -> 128x64 tile,
// BK=32.  Double-buffered LDS filled by async global->LDS B128 copies.
// EPI 0: M1out=C (f32), zout = bf16(C*(1-t) + sig*eps_h)
// EPI 1: outb = bf16(tanh(C))
// EPI 2: outb = bf16(C)
// EPI 3: loss += ((M1in - C) * (1 + w(1-t)/sig) / g)^2   (block-reduced)
// ---------------------------------------------------------------------------
#define AST 80              // LDS row stride (bytes): 16B aligned + conflict-free
#define BST 80
#define ABYTES (128 * AST)  // 10240
#define BBYTES (64 * BST)   // 5120

template<int EPI>
__global__ __launch_bounds__(256)
void gemm_bf16_k(const unsigned short* __restrict__ Ab,
                 const unsigned short* __restrict__ BT,
                 int Nrows, int Ktot, int Mtot,
                 const int* __restrict__ gidx, const float* __restrict__ tv,
                 const float* __restrict__ ghv,
                 const float* __restrict__ epsH, float* __restrict__ M1out,
                 unsigned short* __restrict__ zout, unsigned short* __restrict__ outb,
                 const float* __restrict__ M1in, float* __restrict__ lossAcc)
{
  __shared__ __align__(16) unsigned char smem[2*ABYTES + 2*BBYTES];

  int tid  = threadIdx.x;
  int wid  = tid >> 5;
  int lane = tid & 31;
  int brow = blockIdx.x * 128;
  int bcol = blockIdx.y * 64;

  // low 32 bits of a generic LDS pointer == wave-relative LDS byte address
  unsigned int ldsBase = (unsigned int)(unsigned long long)(void*)&smem[0];
  unsigned int aIdx[2] = { 0u, (unsigned)ABYTES };
  unsigned int bIdx[2] = { (unsigned)(2*ABYTES), (unsigned)(2*ABYTES + BBYTES) };

  // staging assignment: A tile 128 rows x 64B, 2 threads/row (32B each);
  //                     B tile  64 rows x 64B, 4 threads/row (16B each)
  int ra = tid >> 1, akh = (tid & 1) * 32;     // byte offset within row
  int rb = tid >> 2, bkh = (tid & 3) * 16;
  int gra = brow + ra; if (gra >= Nrows) gra = Nrows - 1;
  unsigned int aG0 = (unsigned int)gra * (unsigned int)Ktot * 2u + (unsigned int)akh;
  unsigned int bG0 = (unsigned int)(bcol + rb) * (unsigned int)Ktot * 2u + (unsigned int)bkh;
  unsigned int aL  = (unsigned int)(ra * AST + akh);
  unsigned int bL  = (unsigned int)(rb * BST + bkh);

  int steps = Ktot >> 5;        // BK = 32 halfs = 64 bytes per step
  auto issue = [&](int s) {
    int b = s & 1;
    unsigned int ag = aG0 + (unsigned int)(s * 64);
    unsigned int bg = bG0 + (unsigned int)(s * 64);
    async_b128(ldsBase + aIdx[b] + aL,      ag,      (const void*)Ab);
    async_b128(ldsBase + aIdx[b] + aL + 16, ag + 16, (const void*)Ab);
    async_b128(ldsBase + bIdx[b] + bL,      bg,      (const void*)BT);
  };

  int rlo = lane & 15;
  int hi  = (lane >> 4) & 1;
  unsigned int aFragRow = (unsigned int)((wid*16 + rlo) * AST + hi * 16);

  v8f acc[4] = {};
  issue(0);
  for (int s = 0; s < steps; ++s) {
    if (s + 1 < steps) { issue(s + 1); wait_async3(); }
    else               { wait_async0(); }
    __syncthreads();                        // tile s visible to all waves
    int cur = s & 1;
    Frag af, bf4[4];
    {
      const unsigned char* pa = &smem[aIdx[cur] + aFragRow];
      af.q[0] = *(const uint4*)(pa);        // k pairs 0..7  (+8 if hi lane)
      af.q[1] = *(const uint4*)(pa + 32);   // k pairs 16..23 (+8)
    }
#pragma unroll
    for (int ct = 0; ct < 4; ++ct) {
      const unsigned char* pb = &smem[bIdx[cur] + (unsigned)((ct*16 + rlo) * BST + hi * 16)];
      bf4[ct].q[0] = *(const uint4*)(pb);
      bf4[ct].q[1] = *(const uint4*)(pb + 32);
    }
    __syncthreads();                        // all reads done before buffer reuse
#pragma unroll
    for (int ct = 0; ct < 4; ++ct)
      acc[ct] = __builtin_amdgcn_wmma_f32_16x16x32_bf16(
          false, af.v, false, bf4[ct].v, (short)0, acc[ct], false, false);
  }

  // epilogue: VGPR r -> row wid*16 + r + (laneHi?8:0), col ct*16 + (lane&15)
  int radd = (lane & 16) ? 8 : 0;
  float lsum = 0.0f;
#pragma unroll
  for (int r = 0; r < 8; ++r) {
    int grow = brow + wid*16 + r + radd;
    if (grow < Nrows) {
      float tn = 0.f, sig = 1.f, gv = 1.f;
      if (EPI == 0 || EPI == 3) {
        int g = gidx[grow];
        tn  = tv[g];
        sig = 0.1f + 0.9f * tn;
        if (EPI == 3) gv = ghv[g];
      }
#pragma unroll
      for (int ct = 0; ct < 4; ++ct) {
        int gcol = bcol + ct*16 + rlo;
        float c = acc[ct][r];
        size_t o = (size_t)grow * Mtot + gcol;
        if (EPI == 0) {
          float z = c * (1.0f - tn) + sig * epsH[o];
          M1out[o] = c;
          zout[o]  = f2bf(z);
        } else if (EPI == 1) {
          outb[o] = f2bf(tanhf(c));
        } else if (EPI == 2) {
          outb[o] = f2bf(c);
        } else { // EPI == 3
          float w   = 0.9f + 0.5f * gv * gv / sig;
          float fac = (1.0f + w * (1.0f - tn) / sig) / gv;
          float d   = (M1in[o] - c) * fac;
          lsum += d * d;
        }
      }
    }
  }
  if (EPI == 3) {
    __shared__ float red[256];
    red[tid] = lsum;
    __syncthreads();
    for (int s = 128; s > 0; s >>= 1) {
      if (tid < s) red[tid] += red[tid + s];
      __syncthreads();
    }
    if (tid == 0) atomicAdd(lossAcc, red[0]);
  }
}

// ---------------------------------------------------------------------------
// prep: f32 -> bf16 (vectorized), and W transpose+bf16 (W^T stored [M][K])
// ---------------------------------------------------------------------------
__global__ __launch_bounds__(256)
void conv_bf16_k(const float* __restrict__ src, unsigned short* __restrict__ dst, int n4)
{
  int i = blockIdx.x * 256 + threadIdx.x;
  if (i >= n4) return;
  float4 v = ((const float4*)src)[i];
  uint2 o; o.x = pack2(v.x, v.y); o.y = pack2(v.z, v.w);
  ((uint2*)dst)[i] = o;
}

__global__ __launch_bounds__(256)
void prep_w_k(const float* __restrict__ Wmu, const float* __restrict__ Wr1,
              const float* __restrict__ Wr2,
              unsigned short* WmuT, unsigned short* Wr1T, unsigned short* Wr2T)
{
  const int DH = 256, D2 = 512;
  int i = blockIdx.x * 256 + threadIdx.x;
  if (i < DH*DH) {
    int m = i / DH, k = i % DH;
    WmuT[m*DH + k] = f2bf(Wmu[k*DH + m]);
  } else if (i < DH*DH + D2*DH) {
    int j = i - DH*DH; int m = j / DH, k = j % DH;         // W_r1: [256][512] -> T [512][256]
    Wr1T[m*DH + k] = f2bf(Wr1[k*D2 + m]);
  } else if (i < DH*DH + D2*DH + DH*D2) {
    int j = i - DH*DH - D2*DH; int m = j / D2, k = j % D2; // W_r2: [512][256] -> T [256][512]
    Wr2T[m*D2 + k] = f2bf(Wr2[k*DH + m]);
  }
}

__global__ __launch_bounds__(256)
void graph_init_k(const float* tv, const float* gph, const float* gpp,
                  float* gh, float* gp, int B)
{
  int b = blockIdx.x * 256 + threadIdx.x; if (b >= B) return;
  float sph = log1pf(expf(gph[0]));
  float spp = log1pf(expf(gpp[0]));
  gh[b] = 0.1f + sph * tv[b];
  gp[b] = 0.1f + spp * tv[b];
}

// ---------------------------------------------------------------------------
// pos pipeline
// ---------------------------------------------------------------------------
__global__ __launch_bounds__(256)
void pos1_k(const float* pos, const float* eps, const int* gidx, const float* tv,
            float* nInvU, float* nUe, float* ndUe, float* nY,
            float* aCnt, float* aSumUe, float* aSumdUe, float* aSumInvU, float* aSumY,
            int N)
{
  int n = blockIdx.x*256 + threadIdx.x; if (n >= N) return;
  int g = gidx[n]; float tn = tv[g];
  float px=pos[3*n], py=pos[3*n+1], pz=pos[3*n+2];
  float ex=eps[3*n], ey=eps[3*n+1], ez=eps[3*n+2];
  float il = 1.0f / (sqrtf(px*px+py*py+pz*pz) + 1e-6f);
  float nx=px*il, ny=py*il, nz=pz*il;
  float a = 0.2f + 0.8f*tn, b = 0.1f*tn;
  float nd = nx*ex + ny*ey + nz*ez;
  float Uex = a*ex + b*nx*nd, Uey = a*ey + b*ny*nd, Uez = a*ez + b*nz*nd;
  float dex = 0.8f*ex + 0.1f*nx*nd, dey = 0.8f*ey + 0.1f*ny*nd, dez = 0.8f*ez + 0.1f*nz*nd;
  float U[9] = { a + b*nx*nx, b*nx*ny, b*nx*nz,
                 b*ny*nx, a + b*ny*ny, b*ny*nz,
                 b*nz*nx, b*nz*ny, a + b*nz*nz };
  float iU[9]; inv3(U, iU);
  float yx = iU[0]*ex + iU[3]*ey + iU[6]*ez;   // y = invU^T eps
  float yy = iU[1]*ex + iU[4]*ey + iU[7]*ez;
  float yz = iU[2]*ex + iU[5]*ey + iU[8]*ez;
#pragma unroll
  for (int k=0;k<9;++k) nInvU[9*n+k] = iU[k];
  nUe[3*n]=Uex;  nUe[3*n+1]=Uey;  nUe[3*n+2]=Uez;
  ndUe[3*n]=dex; ndUe[3*n+1]=dey; ndUe[3*n+2]=dez;
  nY[3*n]=yx;    nY[3*n+1]=yy;    nY[3*n+2]=yz;
  atomicAdd(&aCnt[g], 1.0f);
  atomicAdd(&aSumUe[3*g],Uex);  atomicAdd(&aSumUe[3*g+1],Uey);  atomicAdd(&aSumUe[3*g+2],Uez);
  atomicAdd(&aSumdUe[3*g],dex); atomicAdd(&aSumdUe[3*g+1],dey); atomicAdd(&aSumdUe[3*g+2],dez);
#pragma unroll
  for (int k=0;k<9;++k) atomicAdd(&aSumInvU[9*g+k], iU[k]);
  atomicAdd(&aSumY[3*g],yx); atomicAdd(&aSumY[3*g+1],yy); atomicAdd(&aSumY[3*g+2],yz);
}

__global__ __launch_bounds__(256)
void graphA_k(const float* aCnt, const float* aSumUe, const float* aSumdUe,
              const float* aSumInvU, float* invV, float* meanUe, float* meandUe, int B)
{
  int b = blockIdx.x*256 + threadIdx.x; if (b >= B) return;
  float ic = 1.0f / aCnt[b];
#pragma unroll
  for (int d=0;d<3;++d) { meanUe[3*b+d]=aSumUe[3*b+d]*ic; meandUe[3*b+d]=aSumdUe[3*b+d]*ic; }
  float iv[9]; inv3(&aSumInvU[9*b], iv);
#pragma unroll
  for (int k=0;k<9;++k) invV[9*b+k]=iv[k];
}

__global__ __launch_bounds__(256)
void pos2_k(const float* pos, const int* gidx, const float* tv,
            const float* nInvU, const float* nUe, const float* ndUe, const float* nY,
            const float* invV, const float* meanUe, const float* meandUe, const float* aSumY,
            float* ndzp, float* nS, float* nPrel, float* nInvU2, float* nNrm2, float* nPosr,
            float* aSumS, float* aSumInvU2, int N)
{
  int n = blockIdx.x*256 + threadIdx.x; if (n >= N) return;
  int g = gidx[n]; float tn = tv[g]; float om = 1.0f - tn;
  float px=pos[3*n], py=pos[3*n+1], pz=pos[3*n+2];
  float zx = px*om + nUe[3*n]   - meanUe[3*g];
  float zy = py*om + nUe[3*n+1] - meanUe[3*g+1];
  float zz = pz*om + nUe[3*n+2] - meanUe[3*g+2];
  float dzx = -px + ndUe[3*n]   - meandUe[3*g];
  float dzy = -py + ndUe[3*n+1] - meandUe[3*g+1];
  float dzz = -pz + ndUe[3*n+2] - meandUe[3*g+2];
  ndzp[3*n]=dzx; ndzp[3*n+1]=dzy; ndzp[3*n+2]=dzz;
  float C[9]; cmat(&invV[9*g], &nInvU[9*n], C);
  float ybx=aSumY[3*g], yby=aSumY[3*g+1], ybz=aSumY[3*g+2];
  float cx = C[0]*ybx + C[3]*yby + C[6]*ybz;          // c = C^T ybar
  float cy = C[1]*ybx + C[4]*yby + C[7]*ybz;
  float cz = C[2]*ybx + C[5]*yby + C[8]*ybz;
  float sx = cx - nY[3*n], sy = cy - nY[3*n+1], sz = cz - nY[3*n+2];
  nS[3*n]=sx; nS[3*n+1]=sy; nS[3*n+2]=sz;
  atomicAdd(&aSumS[3*g],sx); atomicAdd(&aSumS[3*g+1],sy); atomicAdd(&aSumS[3*g+2],sz);
  float rx = 0.9f*zx, ry = 0.9f*zy, rz = 0.9f*zz;     // pos_r
  nPosr[3*n]=rx; nPosr[3*n+1]=ry; nPosr[3*n+2]=rz;
  nPrel[3*n]  = zx - rx*om; nPrel[3*n+1] = zy - ry*om; nPrel[3*n+2] = zz - rz*om;
  float il = 1.0f / (sqrtf(rx*rx+ry*ry+rz*rz) + 1e-6f);
  float nx=rx*il, ny=ry*il, nz=rz*il;
  nNrm2[3*n]=nx; nNrm2[3*n+1]=ny; nNrm2[3*n+2]=nz;
  float a = 0.2f + 0.8f*tn, b = 0.1f*tn;
  float U2[9] = { a + b*nx*nx, b*nx*ny, b*nx*nz,
                  b*ny*nx, a + b*ny*ny, b*ny*nz,
                  b*nz*nx, b*nz*ny, a + b*nz*nz };
  float iU2[9]; inv3(U2, iU2);
#pragma unroll
  for (int k=0;k<9;++k) { nInvU2[9*n+k]=iU2[k]; atomicAdd(&aSumInvU2[9*g+k], iU2[k]); }
}

__global__ __launch_bounds__(256)
void graphB_k(const float* aCnt, const float* aSumS, const float* aSumInvU2,
              float* invV2, float* meanS, int B)
{
  int b = blockIdx.x*256 + threadIdx.x; if (b >= B) return;
  float ic = 1.0f / aCnt[b];
#pragma unroll
  for (int d=0;d<3;++d) meanS[3*b+d]=aSumS[3*b+d]*ic;
  float iv[9]; inv3(&aSumInvU2[9*b], iv);
#pragma unroll
  for (int k=0;k<9;++k) invV2[9*b+k]=iv[k];
}

__global__ __launch_bounds__(256)
void pos3_k(const int* gidx, const float* gp,
            const float* ndzp, const float* nS, const float* nPrel, const float* nInvU2,
            const float* invV2, const float* meanS,
            float* nTgt, float* nC2, float* aSumCp, int N)
{
  int n = blockIdx.x*256 + threadIdx.x; if (n >= N) return;
  int g = gidx[n]; float gv = gp[g]; float hg2 = 0.5f*gv*gv;
#pragma unroll
  for (int d=0;d<3;++d)
    nTgt[3*n+d] = ndzp[3*n+d] - hg2 * (nS[3*n+d] - meanS[3*g+d]);
  float C2[9]; cmat(&invV2[9*g], &nInvU2[9*n], C2);
#pragma unroll
  for (int k=0;k<9;++k) nC2[9*n+k]=C2[k];
  float p0=nPrel[3*n], p1=nPrel[3*n+1], p2=nPrel[3*n+2];
#pragma unroll
  for (int i=0;i<3;++i)
    atomicAdd(&aSumCp[3*g+i], C2[i*3+0]*p0 + C2[i*3+1]*p1 + C2[i*3+2]*p2);
}

__global__ __launch_bounds__(256)
void pos4_k(const int* gidx, const float* nPrel, const float* nInvU2, const float* nNrm2,
            const float* aSumCp, float* nEps2, float* nY2,
            float* aSumY2, float* aSumdU2e, int N)
{
  int n = blockIdx.x*256 + threadIdx.x; if (n >= N) return;
  int g = gidx[n];
  float p0 = nPrel[3*n]   - aSumCp[3*g];
  float p1 = nPrel[3*n+1] - aSumCp[3*g+1];
  float p2 = nPrel[3*n+2] - aSumCp[3*g+2];
  const float* iU = &nInvU2[9*n];
  float e0 = iU[0]*p0 + iU[1]*p1 + iU[2]*p2;
  float e1 = iU[3]*p0 + iU[4]*p1 + iU[5]*p2;
  float e2 = iU[6]*p0 + iU[7]*p1 + iU[8]*p2;
  nEps2[3*n]=e0; nEps2[3*n+1]=e1; nEps2[3*n+2]=e2;
  float y0 = iU[0]*e0 + iU[3]*e1 + iU[6]*e2;   // invU2^T eps2
  float y1 = iU[1]*e0 + iU[4]*e1 + iU[7]*e2;
  float y2v= iU[2]*e0 + iU[5]*e1 + iU[8]*e2;
  nY2[3*n]=y0; nY2[3*n+1]=y1; nY2[3*n+2]=y2v;
  atomicAdd(&aSumY2[3*g],y0); atomicAdd(&aSumY2[3*g+1],y1); atomicAdd(&aSumY2[3*g+2],y2v);
  float nx=nNrm2[3*n], ny=nNrm2[3*n+1], nz=nNrm2[3*n+2];
  float nd = nx*e0 + ny*e1 + nz*e2;
  atomicAdd(&aSumdU2e[3*g],   0.8f*e0 + 0.1f*nx*nd);
  atomicAdd(&aSumdU2e[3*g+1], 0.8f*e1 + 0.1f*ny*nd);
  atomicAdd(&aSumdU2e[3*g+2], 0.8f*e2 + 0.1f*nz*nd);
}

__global__ __launch_bounds__(256)
void pos5_k(const int* gidx, const float* nC2, const float* nY2, const float* aSumY2,
            float* nS2, float* aSumS2, int N)
{
  int n = blockIdx.x*256 + threadIdx.x; if (n >= N) return;
  int g = gidx[n];
  const float* C2 = &nC2[9*n];
  float ybx=aSumY2[3*g], yby=aSumY2[3*g+1], ybz=aSumY2[3*g+2];
  float s0 = C2[0]*ybx + C2[3]*yby + C2[6]*ybz - nY2[3*n];
  float s1 = C2[1]*ybx + C2[4]*yby + C2[7]*ybz - nY2[3*n+1];
  float s2 = C2[2]*ybx + C2[5]*yby + C2[8]*ybz - nY2[3*n+2];
  nS2[3*n]=s0; nS2[3*n+1]=s1; nS2[3*n+2]=s2;
  atomicAdd(&aSumS2[3*g],s0); atomicAdd(&aSumS2[3*g+1],s1); atomicAdd(&aSumS2[3*g+2],s2);
}

__global__ __launch_bounds__(256)
void graphC_k(const float* aCnt, const float* aSumdU2e, const float* aSumS2,
              float* meandU2e, float* meanS2, int B)
{
  int b = blockIdx.x*256 + threadIdx.x; if (b >= B) return;
  float ic = 1.0f / aCnt[b];
#pragma unroll
  for (int d=0;d<3;++d) { meandU2e[3*b+d]=aSumdU2e[3*b+d]*ic; meanS2[3*b+d]=aSumS2[3*b+d]*ic; }
}

__global__ __launch_bounds__(256)
void pos6_k(const int* gidx, const float* gp,
            const float* nPosr, const float* nNrm2, const float* nEps2,
            const float* nS2, const float* nTgt,
            const float* meandU2e, const float* meanS2,
            float* lossP, int N)
{
  int n = blockIdx.x*256 + threadIdx.x;
  float lsum = 0.0f;
  if (n < N) {
    int g = gidx[n]; float gv = gp[g]; float hg2 = 0.5f*gv*gv; float ig = 1.0f/gv;
    float nx=nNrm2[3*n], ny=nNrm2[3*n+1], nz=nNrm2[3*n+2];
    float e0=nEps2[3*n], e1=nEps2[3*n+1], e2=nEps2[3*n+2];
    float nd = nx*e0 + ny*e1 + nz*e2;
    float du0 = 0.8f*e0 + 0.1f*nx*nd;
    float du1 = 0.8f*e1 + 0.1f*ny*nd;
    float du2 = 0.8f*e2 + 0.1f*nz*nd;
    float dz0 = -nPosr[3*n]   + du0 - meandU2e[3*g];
    float dz1 = -nPosr[3*n+1] + du1 - meandU2e[3*g+1];
    float dz2 = -nPosr[3*n+2] + du2 - meandU2e[3*g+2];
    float a0 = dz0 - hg2*(nS2[3*n]   - meanS2[3*g]);
    float a1 = dz1 - hg2*(nS2[3*n+1] - meanS2[3*g+1]);
    float a2 = dz2 - hg2*(nS2[3*n+2] - meanS2[3*g+2]);
    float d0 = (a0 - nTgt[3*n])   * ig;
    float d1 = (a1 - nTgt[3*n+1]) * ig;
    float d2 = (a2 - nTgt[3*n+2]) * ig;
    lsum = d0*d0 + d1*d1 + d2*d2;
  }
  __shared__ float red[256];
  red[threadIdx.x] = lsum;
  __syncthreads();
  for (int s = 128; s > 0; s >>= 1) {
    if (threadIdx.x < s) red[threadIdx.x] += red[threadIdx.x + s];
    __syncthreads();
  }
  if (threadIdx.x == 0) atomicAdd(lossP, red[0]);
}

__global__ void finalize_k(const float* lossH, const float* lossP, float* out,
                           float invNH, float invNP)
{
  if (threadIdx.x == 0) { out[0] = lossH[0]*invNH; out[1] = lossP[0]*invNP; }
}

// ---------------------------------------------------------------------------
extern "C" void kernel_launch(void* const* d_in, const int* in_sizes, int n_in,
                              void* d_out, int out_size, void* d_ws, size_t ws_size,
                              hipStream_t stream)
{
  const float* tv    = (const float*)d_in[0];
  const float* hN    = (const float*)d_in[1];
  const float* posN  = (const float*)d_in[2];
  const float* epsH  = (const float*)d_in[3];
  const float* epsP  = (const float*)d_in[4];
  const float* gph   = (const float*)d_in[5];
  const float* gpp   = (const float*)d_in[6];
  const float* Wmu   = (const float*)d_in[7];
  const float* Wr1   = (const float*)d_in[8];
  const float* Wr2   = (const float*)d_in[9];
  const int*   gidx  = (const int*)  d_in[10];
  int B = in_sizes[0];
  int N = in_sizes[10];
  const int DH = 256, D2 = 512;
  float* out = (float*)d_out;

  char* wsb = (char*)d_ws; size_t off = 0;
  auto alloc = [&](size_t bytes) -> char* {
    char* p = wsb + off; off += (bytes + 255) & ~(size_t)255; return p;
  };
  unsigned short* WmuT = (unsigned short*)alloc((size_t)DH*DH*2);
  unsigned short* Wr1T = (unsigned short*)alloc((size_t)D2*DH*2);
  unsigned short* Wr2T = (unsigned short*)alloc((size_t)DH*D2*2);
  unsigned short* hbf  = (unsigned short*)alloc((size_t)N*DH*2);  // bf16 copy of h
  unsigned short* zbuf = (unsigned short*)alloc((size_t)N*DH*2);  // z_h, reused as h_r
  float*          M1   = (float*)alloc((size_t)N*DH*4);
  unsigned short* A1   = (unsigned short*)alloc((size_t)N*D2*2);
  float* nInvU  = (float*)alloc((size_t)N*9*4);
  float* nUe    = (float*)alloc((size_t)N*3*4);
  float* ndUe   = (float*)alloc((size_t)N*3*4);
  float* nY     = (float*)alloc((size_t)N*3*4);
  float* ndzp   = (float*)alloc((size_t)N*3*4);
  float* nS     = (float*)alloc((size_t)N*3*4);
  float* nPrel  = (float*)alloc((size_t)N*3*4);
  float* nInvU2 = (float*)alloc((size_t)N*9*4);
  float* nNrm2  = (float*)alloc((size_t)N*3*4);
  float* nPosr  = (float*)alloc((size_t)N*3*4);
  float* nTgt   = (float*)alloc((size_t)N*3*4);
  float* nC2    = (float*)alloc((size_t)N*9*4);
  float* nEps2  = (float*)alloc((size_t)N*3*4);
  float* nY2    = (float*)alloc((size_t)N*3*4);
  float* nS2    = (float*)alloc((size_t)N*3*4);
  float* ghArr  = (float*)alloc((size_t)B*4);
  float* gpArr  = (float*)alloc((size_t)B*4);
  float* invV     = (float*)alloc((size_t)B*9*4);
  float* meanUe   = (float*)alloc((size_t)B*3*4);
  float* meandUe  = (float*)alloc((size_t)B*3*4);
  float* invV2    = (float*)alloc((size_t)B*9*4);
  float* meanS    = (float*)alloc((size_t)B*3*4);
  float* meandU2e = (float*)alloc((size_t)B*3*4);
  float* meanS2   = (float*)alloc((size_t)B*3*4);
  // zeroed accumulator region (contiguous)
  size_t accFloats = (size_t)43*B + 2;
  float* accR = (float*)alloc(accFloats*4);
  float* aCnt     = accR;
  float* aSumUe   = aCnt + B;
  float* aSumdUe  = aSumUe + 3*B;
  float* aSumInvU = aSumdUe + 3*B;
  float* aSumY    = aSumInvU + 9*B;
  float* aSumS    = aSumY + 3*B;
  float* aSumInvU2= aSumS + 3*B;
  float* aSumCp   = aSumInvU2 + 9*B;
  float* aSumY2   = aSumCp + 3*B;
  float* aSumdU2e = aSumY2 + 3*B;
  float* aSumS2   = aSumdU2e + 3*B;
  float* lossH    = aSumS2 + 3*B;
  float* lossP    = lossH + 1;

  hipMemsetAsync(accR, 0, accFloats*4, stream);

  int totW = DH*DH + D2*DH + DH*D2;
  prep_w_k<<<(totW + 255)/256, 256, 0, stream>>>(Wmu, Wr1, Wr2, WmuT, Wr1T, Wr2T);
  graph_init_k<<<(B + 255)/256, 256, 0, stream>>>(tv, gph, gpp, ghArr, gpArr, B);
  int n4 = (N * DH) / 4;
  conv_bf16_k<<<(n4 + 255)/256, 256, 0, stream>>>(hN, hbf, n4);

  int gx  = (N + 127)/128;
  int pgx = (N + 255)/256;
  int ggx = (B + 255)/256;

  // ---- h pipeline ----
  gemm_bf16_k<0><<<dim3(gx, DH/64), 256, 0, stream>>>(
      hbf, WmuT, N, DH, DH, gidx, tv, nullptr, epsH, M1, zbuf,
      nullptr, nullptr, nullptr);
  gemm_bf16_k<1><<<dim3(gx, D2/64), 256, 0, stream>>>(
      zbuf, Wr1T, N, DH, D2, nullptr, nullptr, nullptr, nullptr,
      nullptr, nullptr, A1, nullptr, nullptr);
  gemm_bf16_k<2><<<dim3(gx, DH/64), 256, 0, stream>>>(
      A1, Wr2T, N, D2, DH, nullptr, nullptr, nullptr, nullptr,
      nullptr, nullptr, zbuf /* h_r reuses z buffer */, nullptr, nullptr);
  gemm_bf16_k<3><<<dim3(gx, DH/64), 256, 0, stream>>>(
      zbuf, WmuT, N, DH, DH, gidx, tv, ghArr, nullptr,
      nullptr, nullptr, nullptr, M1, lossH);

  // ---- pos pipeline ----
  pos1_k<<<pgx, 256, 0, stream>>>(posN, epsP, gidx, tv, nInvU, nUe, ndUe, nY,
                                  aCnt, aSumUe, aSumdUe, aSumInvU, aSumY, N);
  graphA_k<<<ggx, 256, 0, stream>>>(aCnt, aSumUe, aSumdUe, aSumInvU,
                                    invV, meanUe, meandUe, B);
  pos2_k<<<pgx, 256, 0, stream>>>(posN, gidx, tv, nInvU, nUe, ndUe, nY,
                                  invV, meanUe, meandUe, aSumY,
                                  ndzp, nS, nPrel, nInvU2, nNrm2, nPosr,
                                  aSumS, aSumInvU2, N);
  graphB_k<<<ggx, 256, 0, stream>>>(aCnt, aSumS, aSumInvU2, invV2, meanS, B);
  pos3_k<<<pgx, 256, 0, stream>>>(gidx, gpArr, ndzp, nS, nPrel, nInvU2,
                                  invV2, meanS, nTgt, nC2, aSumCp, N);
  pos4_k<<<pgx, 256, 0, stream>>>(gidx, nPrel, nInvU2, nNrm2, aSumCp,
                                  nEps2, nY2, aSumY2, aSumdU2e, N);
  pos5_k<<<pgx, 256, 0, stream>>>(gidx, nC2, nY2, aSumY2, nS2, aSumS2, N);
  graphC_k<<<ggx, 256, 0, stream>>>(aCnt, aSumdU2e, aSumS2, meandU2e, meanS2, B);
  pos6_k<<<pgx, 256, 0, stream>>>(gidx, gpArr, nPosr, nNrm2, nEps2, nS2, nTgt,
                                  meandU2e, meanS2, lossP, N);

  finalize_k<<<1, 32, 0, stream>>>(lossH, lossP, out,
                                   1.0f/((float)N*(float)DH), 1.0f/((float)N*3.0f));
}
